// SelfAttention_87866440941711
// MI455X (gfx1250) — matrix operand order repaired and checked by reading, not compiled, not run on hardware
//
#include <hip/hip_runtime.h>
#include <math.h>

// ---------------------------------------------------------------------------
// Transformer block (LN -> QKV -> MHA -> proj+res -> LN -> MLP+res) on CDNA5
// All GEMMs + attention matmuls run on v_wmma_f32_16x16x32_f16 (wave32 WMMA).
// ---------------------------------------------------------------------------

typedef __attribute__((ext_vector_type(16))) _Float16 v16h;
typedef __attribute__((ext_vector_type(8)))  _Float16 v8h;
typedef __attribute__((ext_vector_type(8)))  float    v8f;

#define B_   8
#define N_   2048
#define C_   256
#define H_   4
#define D_   64
#define M_   (B_ * N_)   // 16384 rows

static __device__ __forceinline__ v8f wmma_ff(v16h a, v16h b, v8f c) {
  // D = A(16x32 f16) * B(32x16 f16) + C(16x16 f32)
  return __builtin_amdgcn_wmma_f32_16x16x32_f16(false, a, false, b, (short)0, c,
                                                false, false);
}

// A fragment: 16x32 f16, row-major in memory with leading dim ld.
// Lane l<16 : row=l,      K = 0..15   (contiguous)
// Lane l>=16: row=l-16,   K = 16..31  (contiguous)
static __device__ __forceinline__ v16h load_afrag(const _Float16* base, int ld,
                                                  int lane) {
  const int half = lane >> 4;
  const int r    = lane & 15;
  const _Float16* p = base + (size_t)r * ld + half * 16;
  return *(const v16h*)(p);
}

// B fragment: 32x16 f16 where memory holds B transposed: Bt[col][k], ld = ldk.
// Lane l<16 : col=l,      K = 0..15   (contiguous)
// Lane l>=16: col=l-16,   K = 16..31  (contiguous)
static __device__ __forceinline__ v16h load_bfrag(const _Float16* baseT, int ldk,
                                                  int lane) {
  const int half = lane >> 4;
  const int c    = lane & 15;
  return *(const v16h*)(baseT + (size_t)c * ldk + half * 16);
}

// reduce across the 16 lanes of each wave-half (C-fragment row reduction)
static __device__ __forceinline__ float hmax16(float v) {
#pragma unroll
  for (int m = 1; m < 16; m <<= 1) v = fmaxf(v, __shfl_xor(v, m, 32));
  return v;
}
static __device__ __forceinline__ float hsum16(float v) {
#pragma unroll
  for (int m = 1; m < 16; m <<= 1) v += __shfl_xor(v, m, 32);
  return v;
}

// hard fence: make same-wave LDS stores visible to subsequent cross-lane loads
static __device__ __forceinline__ void lds_fence() {
  __builtin_amdgcn_wave_barrier();
  asm volatile("s_wait_dscnt 0" ::: "memory");
  __builtin_amdgcn_wave_barrier();
}
// compiler-only ordering fence (hardware keeps per-wave DS ops in order)
static __device__ __forceinline__ void sched_fence() {
  __builtin_amdgcn_wave_barrier();
  asm volatile("" ::: "memory");
}

// ---------------------------------------------------------------------------
// Weight convert + transpose: W[K][N] f32  ->  Wt[N][K] f16
// ---------------------------------------------------------------------------
__global__ void wcvt_kernel(const float* __restrict__ W, _Float16* __restrict__ Wt,
                            int K, int N) {
  int idx = blockIdx.x * blockDim.x + threadIdx.x;
  if (idx >= K * N) return;
  int k = idx % K;
  int n = idx / K;
  Wt[(size_t)n * K + k] = (_Float16)W[(size_t)k * N + n];
}

// ---------------------------------------------------------------------------
// LayerNorm(256) fused with f32 -> f16 cast.  One wave per row.
// ---------------------------------------------------------------------------
__global__ void __launch_bounds__(128) ln_kernel(const float* __restrict__ x,
                                                 const float* __restrict__ g,
                                                 const float* __restrict__ be,
                                                 _Float16* __restrict__ out,
                                                 int M) {
  const int lane = threadIdx.x & 31;
  const int row  = blockIdx.x * 4 + (threadIdx.x >> 5);
  if (row >= M) return;
  const float* xr = x + (size_t)row * C_;
  float v[8];
#pragma unroll
  for (int j = 0; j < 8; ++j) v[j] = xr[lane * 8 + j];
  float sm = 0.f, sq = 0.f;
#pragma unroll
  for (int j = 0; j < 8; ++j) { sm += v[j]; sq += v[j] * v[j]; }
#pragma unroll
  for (int m = 1; m < 32; m <<= 1) {
    sm += __shfl_xor(sm, m, 32);
    sq += __shfl_xor(sq, m, 32);
  }
  const float mu  = sm * (1.0f / C_);
  const float var = sq * (1.0f / C_) - mu * mu;
  const float rs  = rsqrtf(var + 1e-5f);
  _Float16* orow = out + (size_t)row * C_;
#pragma unroll
  for (int j = 0; j < 8; ++j) {
    int i = lane * 8 + j;
    orow[i] = (_Float16)((v[j] - mu) * rs * g[i] + be[i]);
  }
}

// ---------------------------------------------------------------------------
// Generic WMMA GEMM: out[M][N] = A[M][K](f16) * Bt[N][K](f16)^T + bias
// One wave computes a 16x128 tile (8 accumulators; A-frag amortized 8x).
// EPI: 0 = bias -> f16 out
//      1 = bias + residual(f32,[M][N]) -> f32 out
//      2 = bias + exact GELU -> f16 out
// ---------------------------------------------------------------------------
template <int EPI>
__global__ void __launch_bounds__(128) gemm_kernel(
    const _Float16* __restrict__ A, const _Float16* __restrict__ Bt,
    const float* __restrict__ bias, const float* __restrict__ resid,
    void* __restrict__ outp, int M, int N, int K) {
  const int lane = threadIdx.x & 31;
  const int wid  = blockIdx.x * 4 + (threadIdx.x >> 5);
  const int tnc  = N >> 7;
  const int total = (M >> 4) * tnc;
  if (wid >= total) return;
  const int tm = wid / tnc;
  const int tn = wid % tnc;
  const int half = lane >> 4;
  const int cc   = lane & 15;

  v8f acc[8];
#pragma unroll
  for (int j = 0; j < 8; ++j) {
    const float bj = bias[tn * 128 + j * 16 + cc];
#pragma unroll
    for (int r = 0; r < 8; ++r) acc[j][r] = bj;
  }

  const _Float16* Abase = A + (size_t)tm * 16 * K;
  for (int kk = 0; kk < K; kk += 32) {
    v16h af = load_afrag(Abase + kk, K, lane);
#pragma unroll
    for (int j = 0; j < 8; ++j) {
      v16h bf = load_bfrag(Bt + (size_t)(tn * 128 + j * 16) * K + kk, K, lane);
      acc[j] = wmma_ff(af, bf, acc[j]);
    }
  }

#pragma unroll
  for (int j = 0; j < 8; ++j) {
    const int col = tn * 128 + j * 16 + cc;
#pragma unroll
    for (int r = 0; r < 8; ++r) {
      const int row = tm * 16 + half * 8 + r;
      float v = acc[j][r];
      if (EPI == 1) {
        v += resid[(size_t)row * N + col];
        ((float*)outp)[(size_t)row * N + col] = v;
      } else if (EPI == 2) {
        v = 0.5f * v * (1.0f + erff(v * 0.70710678118654752f));
        ((_Float16*)outp)[(size_t)row * N + col] = (_Float16)v;
      } else {
        ((_Float16*)outp)[(size_t)row * N + col] = (_Float16)v;
      }
    }
  }
}

// ---------------------------------------------------------------------------
// V transpose: qkv16[b*N+n][512 + h*64 + d] -> vt[(b*H+h)*64 + d][n]
// ---------------------------------------------------------------------------
__global__ void vtrans_kernel(const _Float16* __restrict__ qkv,
                              _Float16* __restrict__ vt) {
  int idx = blockIdx.x * blockDim.x + threadIdx.x;  // B*H*D*N
  if (idx >= B_ * H_ * D_ * N_) return;
  const int n    = idx & (N_ - 1);
  const int rest = idx >> 11;
  const int d = rest & 63;
  const int h = (rest >> 6) & 3;
  const int b = rest >> 8;
  vt[idx] = qkv[((size_t)(b * N_ + n)) * 768 + 512 + h * 64 + d];
}

// ---------------------------------------------------------------------------
// Flash attention: one wave per (b, h, 16-query tile).  D=64, key step 32.
// 1/sqrt(D) = 0.125 is folded into the Q fragments (exact in f16).
// ---------------------------------------------------------------------------
__global__ void __launch_bounds__(128) attn_kernel(
    const _Float16* __restrict__ qkv, const _Float16* __restrict__ vt,
    _Float16* __restrict__ o16) {
  __shared__ _Float16 lds[4 * 16 * 32];
  const int lane = threadIdx.x & 31;
  const int lw   = threadIdx.x >> 5;
  _Float16* P = lds + lw * (16 * 32);

  const int wid = blockIdx.x * 4 + lw;  // B_*H_*(N_/16) = 4096 waves
  const int qt = wid & 127;
  const int h  = (wid >> 7) & 3;
  const int b  = wid >> 9;
  const int half = lane >> 4;
  const int cc   = lane & 15;

  // Q fragments (16 rows x 64 d), pre-scaled by 1/sqrt(D)
  const _Float16* qbase = qkv + ((size_t)(b * N_ + qt * 16)) * 768 + h * 64;
  v16h qA0 = load_afrag(qbase, 768, lane);
  v16h qA1 = load_afrag(qbase + 32, 768, lane);
#pragma unroll
  for (int i = 0; i < 16; ++i) {
    qA0[i] *= (_Float16)0.125f;
    qA1[i] *= (_Float16)0.125f;
  }

  float m[8], s[8];
  v8f o[4];
#pragma unroll
  for (int r = 0; r < 8; ++r) { m[r] = -1e30f; s[r] = 0.0f; }
#pragma unroll
  for (int j = 0; j < 4; ++j)
#pragma unroll
    for (int r = 0; r < 8; ++r) o[j][r] = 0.0f;

  const _Float16* vhead = vt + ((size_t)(b * H_ + h) * 64) * N_;

  for (int kt = 0; kt < N_ / 32; ++kt) {
    const _Float16* kbase =
        qkv + ((size_t)(b * N_ + kt * 32)) * 768 + 256 + h * 64;

    v8f s0, s1;
#pragma unroll
    for (int r = 0; r < 8; ++r) { s0[r] = 0.0f; s1[r] = 0.0f; }

    // S = (Q/8) * K^T  (keys kt*32 .. kt*32+31)
    {
      v16h kb;
      kb = load_bfrag(kbase, 768, lane);
      s0 = wmma_ff(qA0, kb, s0);
      kb = load_bfrag(kbase + 32, 768, lane);
      s0 = wmma_ff(qA1, kb, s0);
      kb = load_bfrag(kbase + (size_t)16 * 768, 768, lane);
      s1 = wmma_ff(qA0, kb, s1);
      kb = load_bfrag(kbase + (size_t)16 * 768 + 32, 768, lane);
      s1 = wmma_ff(qA1, kb, s1);
    }

    // online softmax (row = half*8 + r, col = cc)
#pragma unroll
    for (int r = 0; r < 8; ++r) {
      const float v0 = s0[r];
      const float v1 = s1[r];
      float t = hmax16(fmaxf(v0, v1));
      const float mn = fmaxf(m[r], t);
      const float al = __expf(m[r] - mn);
      const float p0 = __expf(v0 - mn);
      const float p1 = __expf(v1 - mn);
      const float ls = hsum16(p0 + p1);
      s[r] = s[r] * al + ls;
      m[r] = mn;
#pragma unroll
      for (int j = 0; j < 4; ++j) o[j][r] *= al;
      P[(half * 8 + r) * 32 + cc]      = (_Float16)p0;
      P[(half * 8 + r) * 32 + 16 + cc] = (_Float16)p1;
    }
    lds_fence();

    // O += P * V
    const v16h pA = load_afrag(P, 32, lane);
#pragma unroll
    for (int j = 0; j < 4; ++j) {
      v16h vb = load_bfrag(vhead + (size_t)(j * 16) * N_ + kt * 32, N_, lane);
      o[j] = wmma_ff(pA, vb, o[j]);
    }
    sched_fence();  // DS ops are in-order per wave; only block compiler motion
  }

  // normalize and store (f16) into o16[b*N + n][h*64 + d]
  _Float16* obase = o16 + ((size_t)(b * N_ + qt * 16)) * C_ + h * 64;
#pragma unroll
  for (int r = 0; r < 8; ++r) {
    const float inv = 1.0f / s[r];
#pragma unroll
    for (int j = 0; j < 4; ++j) {
      obase[(size_t)(half * 8 + r) * C_ + j * 16 + cc] =
          (_Float16)(o[j][r] * inv);
    }
  }
}

// ---------------------------------------------------------------------------
// Host launcher
// ---------------------------------------------------------------------------
extern "C" void kernel_launch(void* const* d_in, const int* in_sizes, int n_in,
                              void* d_out, int out_size, void* d_ws,
                              size_t ws_size, hipStream_t stream) {
  (void)in_sizes; (void)n_in; (void)out_size; (void)ws_size;

  const float* x      = (const float*)d_in[0];
  const float* qkv_w  = (const float*)d_in[1];
  const float* qkv_b  = (const float*)d_in[2];
  const float* proj_w = (const float*)d_in[3];
  const float* proj_b = (const float*)d_in[4];
  const float* ln1_g  = (const float*)d_in[5];
  const float* ln1_b  = (const float*)d_in[6];
  const float* ln2_g  = (const float*)d_in[7];
  const float* ln2_b  = (const float*)d_in[8];
  const float* w1     = (const float*)d_in[9];
  const float* b1     = (const float*)d_in[10];
  const float* w2     = (const float*)d_in[11];
  const float* b2     = (const float*)d_in[12];
  float* out = (float*)d_out;

  char* ws = (char*)d_ws;
  size_t off = 0;
  auto alloc = [&](size_t bytes) -> void* {
    void* p = ws + off;
    off = (off + bytes + 255) & ~(size_t)255;
    return p;
  };

  _Float16* qkvWt = (_Float16*)alloc((size_t)C_ * 768 * 2);
  _Float16* projWt = (_Float16*)alloc((size_t)C_ * C_ * 2);
  _Float16* w1t = (_Float16*)alloc((size_t)C_ * 512 * 2);
  _Float16* w2t = (_Float16*)alloc((size_t)512 * C_ * 2);
  // bufA: h16 (LN1 out), later reused as attention output o16
  _Float16* bufA = (_Float16*)alloc((size_t)M_ * C_ * 2);
  // bufQKV: qkv f16, later reused as MLP hidden m1 (16384x512 f16)
  _Float16* bufQKV = (_Float16*)alloc((size_t)M_ * 768 * 2);
  // bufVT: V transposed, later reused as h2 (LN2 out)
  _Float16* bufVT = (_Float16*)alloc((size_t)B_ * H_ * D_ * N_ * 2);
  float* x2 = (float*)alloc((size_t)M_ * C_ * 4);

  _Float16* h16  = bufA;
  _Float16* o16  = bufA;
  _Float16* m1   = bufQKV;
  _Float16* h2   = bufVT;

  // 1) weight convert + transpose
  {
    int n;
    n = C_ * 768;  wcvt_kernel<<<(n + 255) / 256, 256, 0, stream>>>(qkv_w, qkvWt, C_, 768);
    n = C_ * C_;   wcvt_kernel<<<(n + 255) / 256, 256, 0, stream>>>(proj_w, projWt, C_, C_);
    n = C_ * 512;  wcvt_kernel<<<(n + 255) / 256, 256, 0, stream>>>(w1, w1t, C_, 512);
    n = 512 * C_;  wcvt_kernel<<<(n + 255) / 256, 256, 0, stream>>>(w2, w2t, 512, C_);
  }

  // 2) LN1 -> h16
  ln_kernel<<<M_ / 4, 128, 0, stream>>>(x, ln1_g, ln1_b, h16, M_);

  // 3) QKV GEMM: [16384,256] x [256,768] -> f16
  gemm_kernel<0><<<(M_ / 16) * (768 / 128) / 4, 128, 0, stream>>>(
      h16, qkvWt, qkv_b, nullptr, (void*)bufQKV, M_, 768, C_);

  // 4) V transpose
  {
    int n = B_ * H_ * D_ * N_;
    vtrans_kernel<<<(n + 255) / 256, 256, 0, stream>>>(bufQKV, bufVT);
  }

  // 5) attention -> o16 (aliases h16, no longer needed)
  attn_kernel<<<(B_ * H_ * (N_ / 16)) / 4, 128, 0, stream>>>(bufQKV, bufVT, o16);

  // 6) proj GEMM + residual(x) -> x2 (f32)
  gemm_kernel<1><<<(M_ / 16) * (C_ / 128) / 4, 128, 0, stream>>>(
      o16, projWt, proj_b, x, (void*)x2, M_, C_, C_);

  // 7) LN2 -> h2 (aliases bufVT)
  ln_kernel<<<M_ / 4, 128, 0, stream>>>(x2, ln2_g, ln2_b, h2, M_);

  // 8) MLP1 GEMM + GELU -> m1 (aliases bufQKV)
  gemm_kernel<2><<<(M_ / 16) * (512 / 128) / 4, 128, 0, stream>>>(
      h2, w1t, b1, nullptr, (void*)m1, M_, 512, C_);

  // 9) MLP2 GEMM + residual(x2) -> out (f32)
  gemm_kernel<1><<<(M_ / 16) * (C_ / 128) / 4, 128, 0, stream>>>(
      m1, w2t, b2, x2, (void*)out, M_, C_, 512);
}
